// NeROShapeRenderer_85684597555768
// MI455X (gfx1250) — matrix operand order repaired and checked by reading, not compiled, not run on hardware
//
#include <hip/hip_runtime.h>
#include <hip/hip_bf16.h>
#include <math.h>

#define R_RAYS 1024
#define NS0    64
#define NSF    128
#define EPAD   64     // embed 39 -> 64
#define L3PAD  224    // 217 -> 224
#define H8PAD  272    // 257 -> 272
#define CINPAD 288    // 265 -> 288

typedef __attribute__((ext_vector_type(16))) _Float16 v16h;
typedef __attribute__((ext_vector_type(8)))  float    v8f;

union FragAB { v16h v; uint4 q[2]; _Float16 h[16]; };
union FragC  { v8f  v; float f[8]; };

enum { ACT_NONE = 0, ACT_SOFTPLUS = 1, ACT_RELU = 2, ACT_SIGMOID = 3, ACT_GRAD = 4 };

__device__ __forceinline__ float sigf(float x) { return 1.f / (1.f + __expf(-x)); }

// ---------------------------------------------------------------------------
// Generic WMMA GEMM:  O = epilogue( A[MxK] * B[KxN] + bias )
// TRANS=0: B(k,n) = W[k*ldw + n]   (contiguous in n)
// TRANS=1: B(k,n) = W[n*ldw + k]   (contiguous in k)
// Tile: 128 x (NT*16) per workgroup, 8 waves, wave = 16 rows x NT*16 cols.
// NT compile-time -> branch-free inner loop: 2*NT ds_load_b128 + NT wmma.
// W K-chunk staged transposed in LDS via b128 loads (both orientations).
// ---------------------------------------------------------------------------
template<int NT, int TRANS>
__global__ __launch_bounds__(256)
void wmma_gemm_kernel(const _Float16* __restrict__ A, int lda,
                      const _Float16* __restrict__ W, long long ldw,
                      const float* __restrict__ bias,
                      _Float16* __restrict__ O, int ldo,
                      float* __restrict__ O32, int ldo32,
                      const _Float16* __restrict__ Zin, int ldz,
                      _Float16* __restrict__ Zout,
                      int M, int N, int K, int nbase, int act, int addout, float scale)
{
  __shared__ _Float16 sW[128][48];   // [n][k], 48-half stride keeps 16B alignment
  const int tid     = threadIdx.x;
  const int wave    = tid >> 5;
  const int lane    = tid & 31;
  const int colLane = lane & 15;
  const int laneHi  = lane >> 4;
  const int m0 = blockIdx.x * 128;
  const int n0 = nbase + blockIdx.y * 128;

  FragC acc[NT];
  #pragma unroll
  for (int nt = 0; nt < NT; ++nt) {
    float b = bias ? bias[n0 + nt * 16 + colLane] : 0.f;
    #pragma unroll
    for (int r = 0; r < 8; ++r) acc[nt].f[r] = b;
  }

  // A fragment base: 16x32 f16, lane<16 -> M=lane,K={0..7,16..23}; lane>=16 -> K={8..15,24..31}
  const int am = m0 + wave * 16 + colLane;
  const _Float16* aptr = A + (size_t)am * lda + laneHi * 8;

  for (int k0 = 0; k0 < K; k0 += 32) {
    __syncthreads();
    if (TRANS == 0) {
      // lane = k row (contiguous n); wave = 16-column segment (wave-uniform guard)
      const int k   = lane;
      const int seg = wave;
      if (seg < NT) {
        const _Float16* wp = W + (size_t)(k0 + k) * ldw + (n0 + seg * 16);
        FragAB t;
        t.q[0] = *(const uint4*)(wp);
        t.q[1] = *(const uint4*)(wp + 8);
        #pragma unroll
        for (int j = 0; j < 16; ++j) sW[seg * 16 + j][k] = t.h[j];
      }
    } else {
      // thread = n row (contiguous k) -> vector loads AND vector LDS stores
      const int n  = tid >> 1;
      const int kk = (tid & 1) << 4;
      if (n < NT * 16) {
        const _Float16* wp = W + (size_t)(n0 + n) * ldw + (k0 + kk);
        uint4 q0 = *(const uint4*)(wp);
        uint4 q1 = *(const uint4*)(wp + 8);
        *(uint4*)&sW[n][kk]     = q0;
        *(uint4*)&sW[n][kk + 8] = q1;
      }
    }
    __syncthreads();
    if (k0 + 32 < K) {  // stream next weight chunk toward L2 (global_prefetch_b8)
      const _Float16* pf = (TRANS == 0)
          ? W + (size_t)(k0 + 32 + lane) * ldw + n0
          : W + (size_t)(n0 + (tid & (NT * 16 - 1))) * ldw + (k0 + 32);
      __builtin_prefetch(pf, 0, 1);
    }

    FragAB a;
    a.q[0] = *(const uint4*)(aptr + k0);
    a.q[1] = *(const uint4*)(aptr + k0 + 16);

    FragAB b[NT];   // B 32x16: lane<16 col=lane K={0..7,16..23}, lane>=16 K={8..15,24..31}
    #pragma unroll
    for (int nt = 0; nt < NT; ++nt) {
      const _Float16* bp = &sW[nt * 16 + colLane][laneHi * 8];
      b[nt].q[0] = *(const uint4*)(bp);
      b[nt].q[1] = *(const uint4*)(bp + 16);
    }
    #pragma unroll
    for (int nt = 0; nt < NT; ++nt)
      acc[nt].v = __builtin_amdgcn_wmma_f32_16x16x32_f16(
          false, a.v, false, b[nt].v, (short)0, acc[nt].v, false, false);
  }

  #pragma unroll
  for (int nt = 0; nt < NT; ++nt) {
    const int col = n0 + nt * 16 + colLane;
    #pragma unroll
    for (int r = 0; r < 8; ++r) {
      const int row = m0 + wave * 16 + r + 8 * laneHi;
      const size_t oi = (size_t)row * ldo + col;
      float z = acc[nt].f[r];
      if (Zout) Zout[oi] = (_Float16)z;   // save pre-activation for backward
      float h;
      if (act == ACT_SOFTPLUS) {          // softplus(100 z)/100
        float y = 100.f * z;
        h = (y > 20.f) ? z : log1pf(__expf(y)) * 0.01f;
      } else if (act == ACT_RELU) {
        h = fmaxf(z, 0.f);
      } else if (act == ACT_SIGMOID) {
        h = sigf(z);
      } else if (act == ACT_GRAD) {       // grad chain: (g W^T) * sigmoid(100 z_pre)
        float zp = (float)Zin[(size_t)row * ldz + col];
        h = z * scale * sigf(100.f * zp);
      } else {
        h = z * scale;
      }
      if (O) {
        if (addout) h += (float)O[oi];
        O[oi] = (_Float16)h;
      }
      if (O32) O32[(size_t)row * ldo32 + col] = h;
    }
  }
}

// --------------------------- small helper kernels ---------------------------

__global__ void convert_w_kernel(const float* __restrict__ src, _Float16* __restrict__ dst,
                                 int rows, int cols, int prows, int pcols, int n)
{
  int i = blockIdx.x * blockDim.x + threadIdx.x;
  if (i >= n) return;
  int r = i / pcols, c = i - r * pcols;
  float v = (r < rows && c < cols) ? src[r * cols + c] : 0.f;
  dst[i] = (_Float16)v;
}

__global__ void convert_b_kernel(const float* __restrict__ src, float* __restrict__ dst,
                                 int cols, int n)
{
  int i = blockIdx.x * blockDim.x + threadIdx.x;
  if (i >= n) return;
  dst[i] = (i < cols) ? src[i] : 0.f;
}

__global__ void ray_setup_kernel(const float* __restrict__ ro, const float* __restrict__ rd,
                                 float* __restrict__ ZV)
{
  int r = blockIdx.x * blockDim.x + threadIdx.x;
  if (r >= R_RAYS) return;
  float ox = ro[r*3], oy = ro[r*3+1], oz = ro[r*3+2];
  float dx = rd[r*3], dy = rd[r*3+1], dz = rd[r*3+2];
  float a = dx*dx + dy*dy + dz*dz;
  float b = 2.f * (ox*dx + oy*dy + oz*dz);
  float mid  = -b / (2.f * a);
  float near = fmaxf(mid - 1.f, 0.05f);
  float far  = mid + 1.f;
  for (int s = 0; s < NS0; ++s)
    ZV[r * NSF + s] = near + (far - near) * ((float)s / (float)(NS0 - 1));
}

__global__ void embed_kernel(const float* __restrict__ ro, const float* __restrict__ rd,
                             const float* __restrict__ z, int ldz, int S,
                             float* __restrict__ pts, _Float16* __restrict__ X0, int n)
{
  int idx = blockIdx.x * blockDim.x + threadIdx.x;
  if (idx >= n) return;
  int r = idx / S, s = idx - r * S;
  float zv = z[(size_t)r * ldz + s];
  float p[3];
  #pragma unroll
  for (int c = 0; c < 3; ++c) p[c] = ro[r*3+c] + rd[r*3+c] * zv;
  if (pts) { pts[idx*3] = p[0]; pts[idx*3+1] = p[1]; pts[idx*3+2] = p[2]; }
  _Float16* e = X0 + (size_t)idx * EPAD;
  e[0] = (_Float16)p[0]; e[1] = (_Float16)p[1]; e[2] = (_Float16)p[2];
  float f = 1.f;
  for (int i = 0; i < 6; ++i) {
    #pragma unroll
    for (int c = 0; c < 3; ++c) {
      e[3 + 6*i + c]     = (_Float16)sinf(f * p[c]);
      e[3 + 6*i + 3 + c] = (_Float16)cosf(f * p[c]);
    }
    f *= 2.f;
  }
  for (int j = 39; j < EPAD; ++j) e[j] = (_Float16)0.f;
}

__global__ void skip_concat_kernel(const _Float16* __restrict__ h3, const _Float16* __restrict__ X0,
                                   _Float16* __restrict__ dst, int n)
{
  int idx = blockIdx.x * blockDim.x + threadIdx.x;
  if (idx >= n) return;
  int m = idx >> 8, c = idx & 255;
  const float inv = 0.70710678118f;
  float v = (c < 217) ? (float)h3[(size_t)m * L3PAD + c] * inv
                      : (float)X0[(size_t)m * EPAD + (c - 217)] * inv;
  dst[(size_t)m * 256 + c] = (_Float16)v;
}

__global__ void extract_sdf_kernel(const _Float16* __restrict__ H8, float* __restrict__ dst,
                                   int dstld, int S, int n)
{
  int idx = blockIdx.x * blockDim.x + threadIdx.x;
  if (idx >= n) return;
  int r = idx / S, s = idx - r * S;
  dst[(size_t)r * dstld + s] = (float)H8[(size_t)idx * H8PAD];
}

__device__ __forceinline__ float ray_radius(const float* o, const float* d, float z)
{
  float x = o[0] + d[0]*z, y = o[1] + d[1]*z, w = o[2] + d[2]*z;
  return sqrtf(x*x + y*y + w*w);
}

__global__ void upsample_kernel(const float* __restrict__ ro, const float* __restrict__ rd,
                                const float* __restrict__ ZV, const float* __restrict__ SDF,
                                float* __restrict__ NEWZ, int S, float inv_s)
{
  int r = blockIdx.x * blockDim.x + threadIdx.x;
  if (r >= R_RAYS) return;
  const float* o  = ro + r*3;
  const float* d  = rd + r*3;
  const float* z  = ZV  + (size_t)r * NSF;
  const float* sd = SDF + (size_t)r * NSF;
  float w[NSF], cdf[NSF];
  float T = 1.f, raw_prev = 0.f;
  float rprev = ray_radius(o, d, z[0]);
  for (int i = 0; i < S - 1; ++i) {
    float rnext = ray_radius(o, d, z[i+1]);
    float raw = (sd[i+1] - sd[i]) / (z[i+1] - z[i] + 1e-5f);
    float cv  = fminf(raw, (i == 0) ? 0.f : raw_prev);
    raw_prev = raw;
    cv = fminf(fmaxf(cv, -1000.f), 0.f);
    if (!(rprev < 1.f || rnext < 1.f)) cv = 0.f;
    float mid  = 0.5f * (sd[i] + sd[i+1]);
    float dist = z[i+1] - z[i];
    float pe = mid - cv * dist * 0.5f;
    float ne = mid + cv * dist * 0.5f;
    float pc = sigf(pe * inv_s), nc = sigf(ne * inv_s);
    float alpha = (pc - nc + 1e-5f) / (pc + 1e-5f);
    w[i] = alpha * T;
    T *= (1.f - alpha + 1e-7f);
    rprev = rnext;
  }
  float wsum = 0.f;
  for (int i = 0; i < S - 1; ++i) wsum += w[i] + 1e-5f;
  cdf[0] = 0.f;
  for (int i = 0; i < S - 1; ++i) cdf[i+1] = cdf[i] + (w[i] + 1e-5f) / wsum;
  for (int j = 0; j < 16; ++j) {
    float u = (0.5f + (float)j) / 16.f;
    int ind = S;
    for (int t = 0; t < S; ++t) if (cdf[t] > u) { ind = t; break; }
    int below = max(0, min(ind - 1, S - 1));
    int above = max(0, min(ind,     S - 1));
    float c0 = cdf[below], c1 = cdf[above];
    float b0 = z[below],   b1 = z[above];
    float den = (c1 - c0 < 1e-5f) ? 1.f : (c1 - c0);
    NEWZ[r * 16 + j] = b0 + (u - c0) / den * (b1 - b0);
  }
}

__global__ void merge_kernel(float* __restrict__ ZV, float* __restrict__ SDF,
                             const float* __restrict__ NEWZ, const float* __restrict__ NSDF, int S)
{
  int r = blockIdx.x * blockDim.x + threadIdx.x;
  if (r >= R_RAYS) return;
  float zo[NSF], so[NSF], zn[16], sn[16];
  for (int i = 0; i < S; ++i) { zo[i] = ZV[r*NSF+i]; so[i] = SDF[r*NSF+i]; }
  for (int j = 0; j < 16; ++j) { zn[j] = NEWZ[r*16+j]; sn[j] = NSDF[r*16+j]; }
  int i = 0, j = 0;
  for (int t = 0; t < S + 16; ++t) {
    bool old = (j >= 16) || (i < S && zo[i] <= zn[j]);
    if (old) { ZV[r*NSF+t] = zo[i]; SDF[r*NSF+t] = so[i]; ++i; }
    else     { ZV[r*NSF+t] = zn[j]; SDF[r*NSF+t] = sn[j]; ++j; }
  }
}

__global__ void prep_final_kernel(const float* __restrict__ ZV, float* __restrict__ DISTS,
                                  float* __restrict__ MIDZ)
{
  int idx = blockIdx.x * blockDim.x + threadIdx.x;
  if (idx >= R_RAYS * NSF) return;
  int s = idx & (NSF - 1);
  float z = ZV[idx];
  float dist = (s < NSF - 1) ? ZV[idx + 1] - z : 2.f / 64.f;
  DISTS[idx] = dist;
  MIDZ[idx]  = z + 0.5f * dist;
}

__global__ void p7_init_kernel(const _Float16* __restrict__ W8, const _Float16* __restrict__ Z7,
                               _Float16* __restrict__ P, int n)
{
  int idx = blockIdx.x * blockDim.x + threadIdx.x;
  if (idx >= n) return;
  int c = idx & 255;
  float g = (float)W8[(size_t)c * H8PAD];   // W8[:,0]
  P[idx] = (_Float16)(g * sigf(100.f * (float)Z7[idx]));
}

__global__ void skipgrad_kernel(const _Float16* __restrict__ GX4, const _Float16* __restrict__ Z3,
                                _Float16* __restrict__ P3, _Float16* __restrict__ GE, int n)
{
  int idx = blockIdx.x * blockDim.x + threadIdx.x;
  if (idx >= n) return;
  int m = idx >> 8, c = idx & 255;
  float g = (float)GX4[idx] * 0.70710678118f;
  if (c < 217) {
    P3[(size_t)m * L3PAD + c] = (_Float16)(g * sigf(100.f * (float)Z3[(size_t)m * L3PAD + c]));
  } else {
    if (c < 224) P3[(size_t)m * L3PAD + c] = (_Float16)0.f;
    GE[(size_t)m * EPAD + (c - 217)] = (_Float16)g;   // e-skip gradient
  }
}

__global__ void grad2pts_kernel(const _Float16* __restrict__ GE, const float* __restrict__ PTS,
                                float* __restrict__ GRD, int n)
{
  int idx = blockIdx.x * blockDim.x + threadIdx.x;
  if (idx >= n) return;
  const _Float16* ge = GE + (size_t)idx * EPAD;
  #pragma unroll
  for (int c = 0; c < 3; ++c) {
    float x = PTS[idx*3 + c];
    float g = (float)ge[c];
    float f = 1.f;
    for (int i = 0; i < 6; ++i) {
      g += f * ((float)ge[3 + 6*i + c] * cosf(f*x) - (float)ge[3 + 6*i + 3 + c] * sinf(f*x));
      f *= 2.f;
    }
    GRD[idx*3 + c] = g;
  }
}

__global__ void final_geom_kernel(const float* __restrict__ rd, const float* __restrict__ PTS,
                                  const float* __restrict__ GRD, const _Float16* __restrict__ H8,
                                  const float* __restrict__ DISTS, const float* __restrict__ variance,
                                  float* __restrict__ ALPHA, _Float16* __restrict__ CIN, int n)
{
  int idx = blockIdx.x * blockDim.x + threadIdx.x;
  if (idx >= n) return;
  int r = idx >> 7;
  float inv_s = fminf(fmaxf(__expf(variance[0] * 10.f), 1e-6f), 1e6f);
  float sdf = (float)H8[(size_t)idx * H8PAD];
  float gx = GRD[idx*3], gy = GRD[idx*3+1], gz = GRD[idx*3+2];
  float dx = rd[r*3],    dy = rd[r*3+1],    dz = rd[r*3+2];
  float tc = dx*gx + dy*gy + dz*gz;
  float ic = fminf(tc, 0.f);                 // -relu(-tc)
  float dist = DISTS[idx];
  float en = sdf + ic * dist * 0.5f;
  float ep = sdf - ic * dist * 0.5f;
  float pc = sigf(ep * inv_s), nc = sigf(en * inv_s);
  float alpha = (pc - nc + 1e-5f) / (pc + 1e-5f);
  ALPHA[idx] = fminf(fmaxf(alpha, 0.f), 1.f);
  float gn = fmaxf(sqrtf(gx*gx + gy*gy + gz*gz), 1e-6f);
  _Float16* cin = CIN + (size_t)idx * CINPAD;
  cin[0] = (_Float16)PTS[idx*3]; cin[1] = (_Float16)PTS[idx*3+1]; cin[2] = (_Float16)PTS[idx*3+2];
  cin[3] = (_Float16)dx;         cin[4] = (_Float16)dy;           cin[5] = (_Float16)dz;
  cin[6] = (_Float16)(gx/gn);    cin[7] = (_Float16)(gy/gn);      cin[8] = (_Float16)(gz/gn);
  const _Float16* feat = H8 + (size_t)idx * H8PAD + 1;
  for (int t = 0; t < 256; ++t) cin[9 + t] = feat[t];
  for (int t = 265; t < CINPAD; ++t) cin[t] = (_Float16)0.f;
}

__global__ void weights_kernel(const float* __restrict__ ALPHA, float* __restrict__ WTS)
{
  int r = blockIdx.x * blockDim.x + threadIdx.x;
  if (r >= R_RAYS) return;
  float T = 1.f;
  for (int s = 0; s < NSF; ++s) {
    float a = ALPHA[r*NSF + s];
    WTS[r*NSF + s] = a * T;
    T *= (1.f - a + 1e-7f);
  }
}

__global__ void composite_kernel(const float* __restrict__ WTS, const float* __restrict__ RGBS,
                                 float* __restrict__ out)
{
  int r = blockIdx.x * blockDim.x + threadIdx.x;
  if (r >= R_RAYS) return;
  #pragma unroll
  for (int c = 0; c < 3; ++c) {
    float acc = 0.f;
    for (int s = 0; s < NSF; ++s)
      acc += WTS[r*NSF + s] * RGBS[(size_t)(r*NSF + s) * 16 + c];
    out[r*3 + c] = acc;
  }
}

// --------------------------------- host side --------------------------------

static inline dim3 g1(int n) { return dim3((unsigned)((n + 255) / 256)); }

#define GEMM_ARGS A, lda, W, ldw, bias, O, ldo, O32, ldo32, Zin, ldz, Zout, M, N, K, nb, act, addout, scale

// Launch full 128-wide tiles with NT=8 plus one remainder launch with NT=rem/16.
static inline void gemm(hipStream_t st, const _Float16* A, int lda,
                        const _Float16* W, long long ldw, int trans,
                        const float* bias, _Float16* O, int ldo,
                        float* O32, int ldo32,
                        const _Float16* Zin, int ldz, _Float16* Zout,
                        int M, int N, int K, int act, int addout, float scale)
{
  const int nfull = N / 128;
  const int rem   = N - nfull * 128;
  if (nfull > 0) {
    dim3 grid((unsigned)(M / 128), (unsigned)nfull);
    int nb = 0;
    if (trans) wmma_gemm_kernel<8,1><<<grid, dim3(256), 0, st>>>(GEMM_ARGS);
    else       wmma_gemm_kernel<8,0><<<grid, dim3(256), 0, st>>>(GEMM_ARGS);
  }
  if (rem > 0) {
    dim3 grid((unsigned)(M / 128), 1u);
    const int nb = nfull * 128;
    switch (rem / 16) {
      case 1: if (trans) wmma_gemm_kernel<1,1><<<grid, dim3(256), 0, st>>>(GEMM_ARGS);
              else       wmma_gemm_kernel<1,0><<<grid, dim3(256), 0, st>>>(GEMM_ARGS); break;
      case 2: if (trans) wmma_gemm_kernel<2,1><<<grid, dim3(256), 0, st>>>(GEMM_ARGS);
              else       wmma_gemm_kernel<2,0><<<grid, dim3(256), 0, st>>>(GEMM_ARGS); break;
      case 3: if (trans) wmma_gemm_kernel<3,1><<<grid, dim3(256), 0, st>>>(GEMM_ARGS);
              else       wmma_gemm_kernel<3,0><<<grid, dim3(256), 0, st>>>(GEMM_ARGS); break;
      case 4: if (trans) wmma_gemm_kernel<4,1><<<grid, dim3(256), 0, st>>>(GEMM_ARGS);
              else       wmma_gemm_kernel<4,0><<<grid, dim3(256), 0, st>>>(GEMM_ARGS); break;
      case 5: if (trans) wmma_gemm_kernel<5,1><<<grid, dim3(256), 0, st>>>(GEMM_ARGS);
              else       wmma_gemm_kernel<5,0><<<grid, dim3(256), 0, st>>>(GEMM_ARGS); break;
      case 6: if (trans) wmma_gemm_kernel<6,1><<<grid, dim3(256), 0, st>>>(GEMM_ARGS);
              else       wmma_gemm_kernel<6,0><<<grid, dim3(256), 0, st>>>(GEMM_ARGS); break;
      default:if (trans) wmma_gemm_kernel<7,1><<<grid, dim3(256), 0, st>>>(GEMM_ARGS);
              else       wmma_gemm_kernel<7,0><<<grid, dim3(256), 0, st>>>(GEMM_ARGS); break;
    }
  }
}

extern "C" void kernel_launch(void* const* d_in, const int* in_sizes, int n_in,
                              void* d_out, int out_size, void* d_ws, size_t ws_size,
                              hipStream_t stream)
{
  (void)in_sizes; (void)n_in; (void)out_size; (void)ws_size;
  const float* rays_o = (const float*)d_in[0];
  const float* rays_d = (const float*)d_in[1];
  const float* sw[9]; const float* sb[9];
  for (int l = 0; l < 9; ++l) { sw[l] = (const float*)d_in[2 + 2*l]; sb[l] = (const float*)d_in[3 + 2*l]; }
  const float* cw[3]; const float* cb[3];
  for (int l = 0; l < 3; ++l) { cw[l] = (const float*)d_in[20 + 2*l]; cb[l] = (const float*)d_in[21 + 2*l]; }
  const float* variance = (const float*)d_in[26];

  const int inK[9]  = {39,256,256,256,256,256,256,256,256};
  const int outN[9] = {256,256,256,217,256,256,256,256,257};
  const int KP[9]   = {64,256,256,256,256,256,256,256,256};
  const int NP[9]   = {256,256,256,224,256,256,256,256,272};
  const int cinK[3] = {265,256,256}, coutN[3] = {256,256,3};
  const int CKP[3]  = {288,256,256}, CNP[3]  = {256,256,16};

  const int M0 = R_RAYS * NS0;        // 65536
  const int MU = R_RAYS * 16;         // 16384
  const int MF = R_RAYS * NSF;        // 131072

  size_t off = 0;
  auto alloc = [&](size_t bytes) -> void* {
    off = (off + 255) & ~(size_t)255;
    void* p = (char*)d_ws + off;
    off += bytes;
    return p;
  };

  _Float16* Wh[9]; float* Bb[9];
  for (int l = 0; l < 9; ++l) {
    Wh[l] = (_Float16*)alloc((size_t)KP[l] * NP[l] * 2);
    Bb[l] = (float*)alloc((size_t)NP[l] * 4);
  }
  _Float16* Ch[3]; float* Cbb[3];
  for (int l = 0; l < 3; ++l) {
    Ch[l]  = (_Float16*)alloc((size_t)CKP[l] * CNP[l] * 2);
    Cbb[l] = (float*)alloc((size_t)CNP[l] * 4);
  }

  _Float16* X0 = (_Float16*)alloc((size_t)MF * EPAD * 2);
  _Float16* HA = (_Float16*)alloc((size_t)MF * H8PAD * 2);
  _Float16* HB = (_Float16*)alloc((size_t)MF * H8PAD * 2);
  _Float16* H8 = (_Float16*)alloc((size_t)MF * H8PAD * 2);
  _Float16* Z[8];
  for (int l = 0; l < 8; ++l) Z[l] = (_Float16*)alloc((size_t)MF * 256 * 2);
  _Float16* PW0 = (_Float16*)alloc((size_t)MF * 256 * 2);
  _Float16* PW1 = (_Float16*)alloc((size_t)MF * 256 * 2);
  _Float16* GX4 = (_Float16*)alloc((size_t)MF * 256 * 2);
  _Float16* GE  = (_Float16*)alloc((size_t)MF * EPAD * 2);
  _Float16* CIN = (_Float16*)alloc((size_t)MF * CINPAD * 2);
  float* PTS   = (float*)alloc((size_t)MF * 3 * 4);
  float* GRD   = (float*)alloc((size_t)MF * 3 * 4);
  float* RGBS  = (float*)alloc((size_t)MF * 16 * 4);
  float* ZV    = (float*)alloc((size_t)R_RAYS * NSF * 4);
  float* SDF   = (float*)alloc((size_t)R_RAYS * NSF * 4);
  float* NEWZ  = (float*)alloc((size_t)R_RAYS * 16 * 4);
  float* NSDF  = (float*)alloc((size_t)R_RAYS * 16 * 4);
  float* DISTS = (float*)alloc((size_t)R_RAYS * NSF * 4);
  float* MIDZ  = (float*)alloc((size_t)R_RAYS * NSF * 4);
  float* ALPHA = (float*)alloc((size_t)R_RAYS * NSF * 4);
  float* WTS   = (float*)alloc((size_t)R_RAYS * NSF * 4);

  // ---- weight / bias conversion to padded fp16 / fp32 ----
  for (int l = 0; l < 9; ++l) {
    int n = KP[l] * NP[l];
    convert_w_kernel<<<g1(n), 256, 0, stream>>>(sw[l], Wh[l], inK[l], outN[l], KP[l], NP[l], n);
    convert_b_kernel<<<g1(NP[l]), 256, 0, stream>>>(sb[l], Bb[l], outN[l], NP[l]);
  }
  for (int l = 0; l < 3; ++l) {
    int n = CKP[l] * CNP[l];
    convert_w_kernel<<<g1(n), 256, 0, stream>>>(cw[l], Ch[l], cinK[l], coutN[l], CKP[l], CNP[l], n);
    convert_b_kernel<<<g1(CNP[l]), 256, 0, stream>>>(cb[l], Cbb[l], coutN[l], CNP[l]);
  }

  auto run_sdf_forward = [&](int M, bool save) {
    const _Float16* cur = X0; int curld = EPAD;
    _Float16* ping[2] = {HA, HB}; int pi = 0;
    for (int l = 0; l < 9; ++l) {
      _Float16* out = (l == 8) ? H8 : ping[pi];
      _Float16* zo  = (save && l < 8) ? Z[l] : nullptr;
      int act = (l == 8) ? ACT_NONE : ACT_SOFTPLUS;
      gemm(stream, cur, curld, Wh[l], NP[l], 0, Bb[l], out, NP[l], nullptr, 0,
           nullptr, 0, zo, M, NP[l], KP[l], act, 0, 1.f);
      cur = out; curld = NP[l]; pi ^= 1;
      if (l == 3) {  // skip connection: x4 = concat(h3, e) / sqrt(2)
        skip_concat_kernel<<<g1(M * 256), 256, 0, stream>>>(cur, X0, ping[pi], M * 256);
        cur = ping[pi]; curld = 256; pi ^= 1;
      }
    }
  };

  // ---- coarse samples ----
  ray_setup_kernel<<<g1(R_RAYS), 256, 0, stream>>>(rays_o, rays_d, ZV);
  embed_kernel<<<g1(M0), 256, 0, stream>>>(rays_o, rays_d, ZV, NSF, NS0, nullptr, X0, M0);
  run_sdf_forward(M0, false);
  extract_sdf_kernel<<<g1(M0), 256, 0, stream>>>(H8, SDF, NSF, NS0, M0);

  // ---- 4 importance up-sample rounds ----
  int S = NS0;
  for (int it = 0; it < 4; ++it) {
    float inv_s = 64.f * (float)(1 << it);
    upsample_kernel<<<g1(R_RAYS), 256, 0, stream>>>(rays_o, rays_d, ZV, SDF, NEWZ, S, inv_s);
    embed_kernel<<<g1(MU), 256, 0, stream>>>(rays_o, rays_d, NEWZ, 16, 16, nullptr, X0, MU);
    run_sdf_forward(MU, false);
    extract_sdf_kernel<<<g1(MU), 256, 0, stream>>>(H8, NSDF, 16, 16, MU);
    merge_kernel<<<g1(R_RAYS), 256, 0, stream>>>(ZV, SDF, NEWZ, NSDF, S);
    S += 16;
  }

  // ---- final forward with saved pre-activations ----
  prep_final_kernel<<<g1(R_RAYS * NSF), 256, 0, stream>>>(ZV, DISTS, MIDZ);
  embed_kernel<<<g1(MF), 256, 0, stream>>>(rays_o, rays_d, MIDZ, NSF, NSF, PTS, X0, MF);
  run_sdf_forward(MF, true);

  // ---- input-gradient backward (VJP wrt points) ----
  p7_init_kernel<<<g1(MF * 256), 256, 0, stream>>>(Wh[8], Z[7], PW0, MF * 256);
  gemm(stream, PW0, 256, Wh[7], 256, 1, nullptr, PW1, 256, nullptr, 0, Z[6], 256, nullptr, MF, 256, 256, ACT_GRAD, 0, 1.f);
  gemm(stream, PW1, 256, Wh[6], 256, 1, nullptr, PW0, 256, nullptr, 0, Z[5], 256, nullptr, MF, 256, 256, ACT_GRAD, 0, 1.f);
  gemm(stream, PW0, 256, Wh[5], 256, 1, nullptr, PW1, 256, nullptr, 0, Z[4], 256, nullptr, MF, 256, 256, ACT_GRAD, 0, 1.f);
  gemm(stream, PW1, 256, Wh[4], 256, 1, nullptr, GX4, 256, nullptr, 0, nullptr, 0, nullptr, MF, 256, 256, ACT_NONE, 0, 1.f);
  skipgrad_kernel<<<g1(MF * 256), 256, 0, stream>>>(GX4, Z[3], PW0, GE, MF * 256);
  gemm(stream, PW0, L3PAD, Wh[3], L3PAD, 1, nullptr, PW1, 256, nullptr, 0, Z[2], 256, nullptr, MF, 256, L3PAD, ACT_GRAD, 0, 1.f);
  gemm(stream, PW1, 256, Wh[2], 256, 1, nullptr, PW0, 256, nullptr, 0, Z[1], 256, nullptr, MF, 256, 256, ACT_GRAD, 0, 1.f);
  gemm(stream, PW0, 256, Wh[1], 256, 1, nullptr, PW1, 256, nullptr, 0, Z[0], 256, nullptr, MF, 256, 256, ACT_GRAD, 0, 1.f);
  gemm(stream, PW1, 256, Wh[0], 256, 1, nullptr, GE, EPAD, nullptr, 0, nullptr, 0, nullptr, MF, EPAD, 256, ACT_NONE, 1, 1.f);
  grad2pts_kernel<<<g1(MF), 256, 0, stream>>>(GE, PTS, GRD, MF);

  // ---- rendering weights + color network ----
  final_geom_kernel<<<g1(MF), 256, 0, stream>>>(rays_d, PTS, GRD, H8, DISTS, variance, ALPHA, CIN, MF);
  weights_kernel<<<g1(R_RAYS), 256, 0, stream>>>(ALPHA, WTS);
  gemm(stream, CIN, CINPAD, Ch[0], CNP[0], 0, Cbb[0], HA, 256, nullptr, 0, nullptr, 0, nullptr, MF, 256, CKP[0], ACT_RELU, 0, 1.f);
  gemm(stream, HA, 256, Ch[1], CNP[1], 0, Cbb[1], HB, 256, nullptr, 0, nullptr, 0, nullptr, MF, 256, 256, ACT_RELU, 0, 1.f);
  gemm(stream, HB, 256, Ch[2], CNP[2], 0, Cbb[2], nullptr, 16, RGBS, 16, nullptr, 0, nullptr, MF, 16, 256, ACT_SIGMOID, 0, 1.f);
  composite_kernel<<<g1(R_RAYS), 256, 0, stream>>>(WTS, RGBS, (float*)d_out);
}
